// DeepseekV3Attention_18442589569925
// MI455X (gfx1250) — compile-verified, matrix operand order checked by reading, not run on hardware
//
#include <hip/hip_runtime.h>
#include <math.h>

// ---------------- types / helpers ----------------
typedef __attribute__((ext_vector_type(16))) __bf16 v16bf;
typedef __attribute__((ext_vector_type(8)))  float  v8f;
typedef __attribute__((ext_vector_type(4)))  unsigned int u32x4;
typedef __attribute__((ext_vector_type(8)))  int i32x8;
typedef __attribute__((ext_vector_type(4)))  int i32x4;
typedef unsigned short u16;
typedef unsigned int   u32;
typedef unsigned long long u64;

#define B_    2
#define S_    2048
#define HID_  2048
#define H_    16
#define NOPE_ 128
#define ROPE_ 64
#define VD_   128
#define QHD_  192          // NOPE + ROPE
#define QL_   1536
#define KVL_  512
#define SCALE_ 0.07216878364870323f   // 192^-0.5

union FragU { v16bf v; u32 u[8]; };

__device__ __forceinline__ u16 f2bf(float x) {
  u32 u = __float_as_uint(x);
  u32 r = u + 0x7fffu + ((u >> 16) & 1u);
  return (u16)(r >> 16);
}

// Load a 16x32 bf16 WMMA A-fragment (or the matching B-fragment packing for a
// row-major [N,K] operand): lane&15 selects the row, 8 dword loads per lane.
__device__ __forceinline__ void load_frag(const u16* base, int stride, int lane, FragU& f) {
  int r = lane & 15, hi = lane >> 4;
#pragma unroll
  for (int v = 0; v < 8; ++v) {
    int kb = (v < 4 ? (v << 1) : (16 + ((v - 4) << 1))) + (hi << 3);
    f.u[v] = *(const u32*)(base + (size_t)r * stride + kb);
  }
}

__device__ __forceinline__ v8f wmma_bf(const FragU& a, const FragU& b, v8f c) {
  return __builtin_amdgcn_wmma_f32_16x16x32_bf16(false, a.v, false, b.v, (short)0, c, false, false);
}

__device__ __forceinline__ float redmax16(float x) {
#pragma unroll
  for (int o = 8; o >= 1; o >>= 1) x = fmaxf(x, __shfl_xor(x, o, 32));
  return x;
}
__device__ __forceinline__ float redsum16(float x) {
#pragma unroll
  for (int o = 8; o >= 1; o >>= 1) x += __shfl_xor(x, o, 32);
  return x;
}

// ---------------- Tensor Data Mover: 2D tile (bf16) global -> LDS ----------------
// D# per CDNA5 ISA ch.8: group0 = {count, lds_addr, global_addr[31:0],
// global_addr[56:32] | type<<30}; group1 packs data_size / tensor dims /
// tile dims / dim0 stride. tensor dims == tile dims here (no OOB clipping).
__device__ __forceinline__ void tdm_load_2d_bf16(const void* gptr, u32 lds_byte_off,
                                                 u32 tile_d0, u32 tile_d1,
                                                 u64 stride0_elems) {
  u64 ga = (u64)(uintptr_t)gptr;
  u32x4 g0;
  g0.x = 1u;                                   // count = 1 valid descriptor
  g0.y = lds_byte_off;                         // LDS destination (bytes)
  g0.z = (u32)ga;                              // global_addr[31:0]
  g0.w = ((u32)(ga >> 32) & 0x01FFFFFFu) | (2u << 30);  // addr[56:32] | type=2
  u32 td0 = tile_d0, td1 = tile_d1;
  i32x8 g1;
  g1[0] = (int)(1u << 16);                               // data_size = 2 bytes
  g1[1] = (int)((td0 & 0xFFFFu) << 16);                  // tensor_dim0[15:0]
  g1[2] = (int)((td0 >> 16) | ((td1 & 0xFFFFu) << 16));  // td0[31:16] | td1[15:0]
  g1[3] = (int)((td1 >> 16) | (tile_d0 << 16));          // td1[31:16] | tile_dim0
  g1[4] = (int)(tile_d1 & 0xFFFFu);                      // tile_dim1 (tile_dim2=0)
  g1[5] = (int)(u32)(stride0_elems & 0xFFFFFFFFu);       // dim0 stride lo
  g1[6] = (int)(u32)(stride0_elems >> 32);               // dim0 stride hi (48b)
  g1[7] = 0;
  i32x4 z4 = {0, 0, 0, 0};
#if defined(__clang_major__) && (__clang_major__ >= 23)
  i32x8 z8 = {0, 0, 0, 0, 0, 0, 0, 0};
  __builtin_amdgcn_tensor_load_to_lds(g0, g1, z4, z4, z8, 0);
#else
  __builtin_amdgcn_tensor_load_to_lds(g0, g1, z4, z4, 0);
#endif
}

// ---------------- elementwise kernels ----------------
__global__ void cvt_f32_bf16(const float* __restrict__ in, u16* __restrict__ out, size_t n) {
  for (size_t i = (size_t)blockIdx.x * blockDim.x + threadIdx.x; i < n;
       i += (size_t)gridDim.x * blockDim.x)
    out[i] = f2bf(in[i]);
}

// one block per row; normalize first n cols of a row with stride `ldin`
__global__ __launch_bounds__(256) void rmsnorm_bf16(const float* __restrict__ in, int ldin,
                                                    int n, const float* __restrict__ w,
                                                    u16* __restrict__ out) {
  int row = blockIdx.x, tid = threadIdx.x;
  const float* x = in + (size_t)row * ldin;
  __shared__ float red[256];
  float s = 0.f;
  for (int i = tid; i < n; i += 256) { float v = x[i]; s += v * v; }
  red[tid] = s; __syncthreads();
  for (int o = 128; o > 0; o >>= 1) { if (tid < o) red[tid] += red[tid + o]; __syncthreads(); }
  float rs = rsqrtf(red[0] / (float)n + 1e-6f);
  for (int i = tid; i < n; i += 256)
    out[(size_t)row * n + i] = f2bf(x[i] * rs * w[i]);
}

__device__ __forceinline__ float rope_val(const float* x, int j, int s) {
  int jj = j & 31;
  float inv = __powf(10000.f, -(float)jj * (1.f / 32.f));
  float ang = (float)s * inv, sn, cs;
  __sincosf(ang, &sn, &cs);
  if (j < 32) return x[2 * j] * cs - x[2 * j + 1] * sn;
  int i2 = j - 32;
  return x[2 * i2 + 1] * cs + x[2 * i2] * sn;
}

// qfull f32 [B*S, H*192] -> query bf16 [B,H,S,192] with RoPE on d in [128,192)
__global__ void build_query(const float* __restrict__ qfull, u16* __restrict__ query) {
  size_t idx = (size_t)blockIdx.x * blockDim.x + threadIdx.x;
  if (idx >= (size_t)B_ * S_ * H_ * QHD_) return;
  int d = (int)(idx % QHD_);
  int h = (int)((idx / QHD_) % H_);
  int row = (int)(idx / ((size_t)QHD_ * H_));
  int s = row & (S_ - 1), b = row >> 11;
  const float* qr = qfull + (size_t)row * (H_ * QHD_) + h * QHD_;
  float val = (d < NOPE_) ? qr[d] : rope_val(qr + NOPE_, d - NOPE_, s);
  query[(((size_t)(b * H_ + h)) * S_ + s) * QHD_ + d] = f2bf(val);
}

// ckv f32 [B*S,576] cols 512..575 -> kpe bf16 [B*S,64] with RoPE
__global__ void build_kpe(const float* __restrict__ ckv, u16* __restrict__ kpe) {
  size_t idx = (size_t)blockIdx.x * blockDim.x + threadIdx.x;
  if (idx >= (size_t)B_ * S_ * ROPE_) return;
  int j = (int)(idx % ROPE_);
  int row = (int)(idx / ROPE_);
  int s = row & (S_ - 1);
  kpe[(size_t)row * ROPE_ + j] = f2bf(rope_val(ckv + (size_t)row * 576 + KVL_, j, s));
}

// kvfull f32 [B*S, H*256] + kpe -> key bf16 [B,H,S,192], value bf16 [B,H,S,128]
__global__ void build_kv(const float* __restrict__ kvfull, const u16* __restrict__ kpe,
                         u16* __restrict__ key, u16* __restrict__ value) {
  size_t idx = (size_t)blockIdx.x * blockDim.x + threadIdx.x;
  if (idx >= (size_t)B_ * S_ * H_ * 320) return;
  int d = (int)(idx % 320);
  int h = (int)((idx / 320) % H_);
  int row = (int)(idx / (320 * H_));
  int s = row & (S_ - 1), b = row >> 11;
  const float* kr = kvfull + (size_t)row * (H_ * 256) + h * 256;
  size_t bhs = ((size_t)(b * H_ + h)) * S_ + s;
  if (d < QHD_) {
    u16 v = (d < NOPE_) ? f2bf(kr[d]) : kpe[(size_t)row * ROPE_ + (d - NOPE_)];
    key[bhs * QHD_ + d] = v;
  } else {
    int dv = d - QHD_;
    value[bhs * VD_ + dv] = f2bf(kr[NOPE_ + dv]);
  }
}

// ---------------- bf16 WMMA GEMM: C[M,N] = A[M,K] * W[N,K]^T ----------------
// block = 256 thr (8 waves as 4x2), block tile 128x64, K step 32.
// TDM DMA's A/W tiles into double-buffered LDS; wave 0 owns the descriptors,
// s_wait_tensorcnt(2) pipelines DMA(k+1) against WMMA(k).
__global__ __launch_bounds__(256) void gemm_bf16(const u16* __restrict__ A,
                                                 const u16* __restrict__ W,
                                                 float* __restrict__ Cf, u16* __restrict__ Cb,
                                                 int M, int N, int K) {
  __shared__ __align__(16) u16 a_lds[2][128 * 32];
  __shared__ __align__(16) u16 b_lds[2][64 * 32];
  int tid = threadIdx.x, lane = tid & 31, w = tid >> 5;
  int mw = w >> 1, nw = w & 1;
  int m0 = blockIdx.y * 128, n0 = blockIdx.x * 64;
  v8f zero = {0.f,0.f,0.f,0.f,0.f,0.f,0.f,0.f};
  v8f acc[2][2] = {{zero, zero}, {zero, zero}};
  int nk = K >> 5;

  u32 a_off[2] = { (u32)(uintptr_t)&a_lds[0][0], (u32)(uintptr_t)&a_lds[1][0] };
  u32 b_off[2] = { (u32)(uintptr_t)&b_lds[0][0], (u32)(uintptr_t)&b_lds[1][0] };

  auto issue = [&](int kt, int buf) {
    const u16* Ak = A + (size_t)m0 * K + kt * 32;
    const u16* Wk = W + (size_t)n0 * K + kt * 32;
    tdm_load_2d_bf16(Ak, a_off[buf], 32u, 128u, (u64)K);
    tdm_load_2d_bf16(Wk, b_off[buf], 32u, 64u,  (u64)K);
  };

  if (w == 0) issue(0, 0);
  for (int kt = 0; kt < nk; ++kt) {
    int buf = kt & 1;
    if (w == 0 && kt + 1 < nk) issue(kt + 1, buf ^ 1);
    if (kt + 1 < nk) __builtin_amdgcn_s_wait_tensorcnt((short)2);
    else             __builtin_amdgcn_s_wait_tensorcnt((short)0);
    __syncthreads();
    FragU af[2], bf[2];
    load_frag(&a_lds[buf][(mw * 32 + 0)  * 32], 32, lane, af[0]);
    load_frag(&a_lds[buf][(mw * 32 + 16) * 32], 32, lane, af[1]);
    load_frag(&b_lds[buf][(nw * 32 + 0)  * 32], 32, lane, bf[0]);
    load_frag(&b_lds[buf][(nw * 32 + 16) * 32], 32, lane, bf[1]);
    acc[0][0] = wmma_bf(af[0], bf[0], acc[0][0]);
    acc[0][1] = wmma_bf(af[0], bf[1], acc[0][1]);
    acc[1][0] = wmma_bf(af[1], bf[0], acc[1][0]);
    acc[1][1] = wmma_bf(af[1], bf[1], acc[1][1]);
    __syncthreads();   // reads done before this buffer's next DMA overwrite
  }
  int n_ = lane & 15, hi = lane >> 4;
#pragma unroll
  for (int mf = 0; mf < 2; ++mf)
#pragma unroll
    for (int nf = 0; nf < 2; ++nf)
#pragma unroll
      for (int v = 0; v < 8; ++v) {
        int gr = m0 + mw * 32 + mf * 16 + v + (hi << 3);
        int gc = n0 + nw * 32 + nf * 16 + n_;
        float val = acc[mf][nf][v];
        if (Cf) Cf[(size_t)gr * N + gc] = val;
        else    Cb[(size_t)gr * N + gc] = f2bf(val);
      }
}

// ---------------- flash attention (bf16 WMMA, online softmax) ----------------
// grid = (S/128, B*H); 8 waves, each owns 16 q-rows. K tiles of 32.
// K tile comes in via TDM; V tile is transposed manually into LDS.
__global__ __launch_bounds__(256) void flash_attn(const u16* __restrict__ query,
                                                  const u16* __restrict__ key,
                                                  const u16* __restrict__ value,
                                                  u16* __restrict__ attnout) {
  __shared__ __align__(16) u16 k_lds[32 * 192];    // 32 k-rows x 192 bf16
  __shared__ __align__(16) u16 vT_lds[128 * 32];   // transposed V: [dv][k]
  __shared__ __align__(16) u16 p_lds[8][16 * 32];  // per-wave P tile

  int qt = blockIdx.x, bh = blockIdx.y;
  int b = bh / H_, h = bh % H_;
  const u16* Qb = query + (size_t)bh * S_ * QHD_;
  const u16* Kb = key   + (size_t)bh * S_ * QHD_;
  const u16* Vb = value + (size_t)bh * S_ * VD_;
  int tid = threadIdx.x, lane = tid & 31, w = tid >> 5;
  int q0 = qt * 128 + w * 16;
  int n_ = lane & 15, hi = lane >> 4;
  u32 k_off = (u32)(uintptr_t)&k_lds[0];

  FragU qf[6];
#pragma unroll
  for (int kk = 0; kk < 6; ++kk)
    load_frag(Qb + (size_t)q0 * QHD_ + kk * 32, QHD_, lane, qf[kk]);

  v8f zero = {0.f,0.f,0.f,0.f,0.f,0.f,0.f,0.f};
  v8f oacc[8];
  float mi[8], li[8];
#pragma unroll
  for (int t = 0; t < 8; ++t) oacc[t] = zero;
#pragma unroll
  for (int v = 0; v < 8; ++v) { mi[v] = -1e30f; li[v] = 0.f; }

  int nkt = qt * 4 + 4;
  for (int kt = 0; kt < nkt; ++kt) {
    int kb0 = kt * 32;
    __syncthreads();
    // K tile (32 rows x 192, contiguous rows) via Tensor Data Mover
    if (w == 0)
      tdm_load_2d_bf16(Kb + (size_t)kb0 * QHD_, k_off, (u32)QHD_, 32u, (u64)QHD_);
    // V tile -> LDS transposed ([dv][k]) by all waves
#pragma unroll
    for (int i = 0; i < 8; ++i) {
      int idx = tid + (i << 8);                    // 0..2047 dwords
      int r = idx >> 6, cu = idx & 63;
      u32 val = *(const u32*)(Vb + (size_t)(kb0 + r) * VD_ + (cu << 1));
      int dv0 = cu << 1;
      vT_lds[dv0 * 32 + r]       = (u16)(val & 0xffffu);
      vT_lds[(dv0 + 1) * 32 + r] = (u16)(val >> 16);
    }
    __builtin_amdgcn_s_wait_tensorcnt((short)0);
    __syncthreads();

    // scores: two 16x16 tiles over the 32 k columns, K-dim = 192 (6 WMMAs each)
    v8f s0 = zero, s1 = zero;
#pragma unroll
    for (int kk = 0; kk < 6; ++kk) {
      FragU kf0, kf1;
      load_frag(&k_lds[0  * QHD_ + kk * 32], QHD_, lane, kf0);
      load_frag(&k_lds[16 * QHD_ + kk * 32], QHD_, lane, kf1);
      s0 = wmma_bf(qf[kk], kf0, s0);
      s1 = wmma_bf(qf[kk], kf1, s1);
    }

    // online softmax per row (row m = v + 8*hi, spread across 16 lanes)
    float p0[8], p1[8];
#pragma unroll
    for (int v = 0; v < 8; ++v) {
      int qpos = q0 + v + (hi << 3);
      float a = s0[v] * SCALE_ + ((kb0 + n_)      > qpos ? -1e9f : 0.f);
      float c = s1[v] * SCALE_ + ((kb0 + 16 + n_) > qpos ? -1e9f : 0.f);
      float rm = redmax16(fmaxf(a, c));
      float mn = fmaxf(mi[v], rm);
      float alpha = __expf(mi[v] - mn);
      float e0 = __expf(a - mn), e1 = __expf(c - mn);
      float rs = redsum16(e0 + e1);
      li[v] = li[v] * alpha + rs;
      mi[v] = mn;
#pragma unroll
      for (int t = 0; t < 8; ++t) oacc[t][v] *= alpha;
      p0[v] = e0; p1[v] = e1;
    }

    // P (C-layout) -> LDS -> A-layout fragment
#pragma unroll
    for (int v = 0; v < 8; ++v) {
      int m = v + (hi << 3);
      p_lds[w][m * 32 + n_]      = f2bf(p0[v]);
      p_lds[w][m * 32 + 16 + n_] = f2bf(p1[v]);
    }
    asm volatile("s_wait_dscnt 0" ::: "memory");
    FragU pf;
    load_frag(&p_lds[w][0], 32, lane, pf);
    // O += P * V  (8 dv tiles of 16)
#pragma unroll
    for (int t = 0; t < 8; ++t) {
      FragU vf;
      load_frag(&vT_lds[(t * 16) * 32], 32, lane, vf);
      oacc[t] = wmma_bf(pf, vf, oacc[t]);
    }
  }

  // epilogue: normalize, write bf16 to [b*S+s, h*VD+dv]
#pragma unroll
  for (int t = 0; t < 8; ++t)
#pragma unroll
    for (int v = 0; v < 8; ++v) {
      int m = v + (hi << 3);
      int r = q0 + m;
      int dv = t * 16 + n_;
      float o = oacc[t][v] / li[v];
      attnout[(size_t)(b * S_ + r) * (H_ * VD_) + h * VD_ + dv] = f2bf(o);
    }
}

// ---------------- host orchestration ----------------
extern "C" void kernel_launch(void* const* d_in, const int* in_sizes, int n_in,
                              void* d_out, int out_size, void* d_ws, size_t ws_size,
                              hipStream_t stream) {
  (void)in_sizes; (void)n_in; (void)out_size; (void)ws_size;
  const float* hidden    = (const float*)d_in[0];
  const float* q_a_w     = (const float*)d_in[3];
  const float* q_a_ln_w  = (const float*)d_in[4];
  const float* q_b_w     = (const float*)d_in[5];
  const float* kv_a_w    = (const float*)d_in[6];
  const float* kv_a_ln_w = (const float*)d_in[7];
  const float* kv_b_w    = (const float*)d_in[8];
  const float* o_w       = (const float*)d_in[9];
  float* out = (float*)d_out;

  const int M = B_ * S_;                       // 4096 token rows
  char* ws = (char*)d_ws;
  size_t off = 0;
  auto alloc = [&](size_t bytes) -> char* {
    char* p = ws + off;
    off = (off + bytes + 255) & ~(size_t)255;
    return p;
  };
  u16* hidden_bf = (u16*)alloc((size_t)M * HID_ * 2);
  u16* wqa  = (u16*)alloc((size_t)QL_ * HID_ * 2);
  u16* wkva = (u16*)alloc((size_t)(KVL_ + ROPE_) * HID_ * 2);
  u16* wqb  = (u16*)alloc((size_t)(H_ * QHD_) * QL_ * 2);
  u16* wkvb = (u16*)alloc((size_t)(H_ * 256) * KVL_ * 2);
  u16* wo   = (u16*)alloc((size_t)HID_ * (H_ * VD_) * 2);
  float* qa_f32  = (float*)alloc((size_t)M * QL_ * 4);
  u16*   qa_n    = (u16*)alloc((size_t)M * QL_ * 2);
  float* ckv_f32 = (float*)alloc((size_t)M * (KVL_ + ROPE_) * 4);
  u16*   ckv_n   = (u16*)alloc((size_t)M * KVL_ * 2);
  float* qfull   = (float*)alloc((size_t)M * (H_ * QHD_) * 4);
  float* kvfull  = (float*)alloc((size_t)M * (H_ * 256) * 4);
  u16*   kpe     = (u16*)alloc((size_t)M * ROPE_ * 2);
  u16*   query   = (u16*)alloc((size_t)B_ * H_ * S_ * QHD_ * 2);
  u16*   keyb    = (u16*)alloc((size_t)B_ * H_ * S_ * QHD_ * 2);
  u16*   valueb  = (u16*)alloc((size_t)B_ * H_ * S_ * VD_ * 2);
  u16*   attnout = (u16*)alloc((size_t)M * (H_ * VD_) * 2);

  auto cvt = [&](const float* in, u16* o2, size_t n) {
    cvt_f32_bf16<<<2048, 256, 0, stream>>>(in, o2, n);
  };
  auto gemm = [&](const u16* A, const u16* W, float* Cf, u16* Cb, int Mm, int Nn, int Kk) {
    dim3 g(Nn / 64, Mm / 128);
    gemm_bf16<<<g, 256, 0, stream>>>(A, W, Cf, Cb, Mm, Nn, Kk);
  };

  // 1) downconvert activations + weights to bf16
  cvt(hidden,  hidden_bf, (size_t)M * HID_);
  cvt(q_a_w,   wqa,  (size_t)QL_ * HID_);
  cvt(kv_a_w,  wkva, (size_t)(KVL_ + ROPE_) * HID_);
  cvt(q_b_w,   wqb,  (size_t)(H_ * QHD_) * QL_);
  cvt(kv_b_w,  wkvb, (size_t)(H_ * 256) * KVL_);
  cvt(o_w,     wo,   (size_t)HID_ * (H_ * VD_));

  // 2) low-rank projections
  gemm(hidden_bf, wqa,  qa_f32,  nullptr, M, QL_,          HID_);   // q_a
  gemm(hidden_bf, wkva, ckv_f32, nullptr, M, KVL_ + ROPE_, HID_);   // kv_a (+k_pe)
  rmsnorm_bf16<<<M, 256, 0, stream>>>(qa_f32, QL_, QL_, q_a_ln_w, qa_n);
  rmsnorm_bf16<<<M, 256, 0, stream>>>(ckv_f32, KVL_ + ROPE_, KVL_, kv_a_ln_w, ckv_n);

  // 3) up projections
  gemm(qa_n,  wqb,  qfull,  nullptr, M, H_ * QHD_, QL_);            // q_b
  gemm(ckv_n, wkvb, kvfull, nullptr, M, H_ * 256,  KVL_);           // kv_b

  // 4) RoPE + head rearrangement
  {
    size_t n = (size_t)M * ROPE_;
    build_kpe<<<(unsigned)((n + 255) / 256), 256, 0, stream>>>(ckv_f32, kpe);
  }
  {
    size_t n = (size_t)M * H_ * QHD_;
    build_query<<<(unsigned)((n + 255) / 256), 256, 0, stream>>>(qfull, query);
  }
  {
    size_t n = (size_t)M * H_ * 320;
    build_kv<<<(unsigned)((n + 255) / 256), 256, 0, stream>>>(kvfull, kpe, keyb, valueb);
  }

  // 5) causal flash attention
  {
    dim3 g(S_ / 128, B_ * H_);
    flash_attn<<<g, 256, 0, stream>>>(query, keyb, valueb, attnout);
  }

  // 6) output projection -> f32 d_out
  gemm(attnout, wo, out, nullptr, M, HID_, H_ * VD_);
}
